// Attention_45938970198347
// MI455X (gfx1250) — matrix operand order repaired and checked by reading, not compiled, not run on hardware
//
#include <hip/hip_runtime.h>

// Problem constants (match reference)
#define B_  2
#define T_  4096
#define C_  1024
#define H_  16
#define D_  64
#define BT_ (B_*T_)

typedef __attribute__((ext_vector_type(16))) __bf16 v16bf;
typedef __attribute__((ext_vector_type(8)))  float  v8f;

union BFrag { v16bf v; uint4 q[2]; };

__device__ __forceinline__ unsigned short f2b(float f) {
    unsigned int u = __builtin_bit_cast(unsigned int, f);
    u += 0x7FFFu + ((u >> 16) & 1u);           // round-to-nearest-even
    return (unsigned short)(u >> 16);
}
__device__ __forceinline__ float b2f(unsigned short s) {
    unsigned int u = ((unsigned int)s) << 16;
    return __builtin_bit_cast(float, u);
}

// ---------------------------------------------------------------------------
// f32 -> bf16 conversion
// ---------------------------------------------------------------------------
__global__ void cvt_f32_bf16(const float* __restrict__ src,
                             unsigned short* __restrict__ dst, int n) {
    int i = blockIdx.x * blockDim.x + threadIdx.x;
    if (i < n) dst[i] = f2b(src[i]);
}

// ---------------------------------------------------------------------------
// GEMM: C[M,N] = A[M,K](bf16) * W[N,K]^T(bf16), f32 accumulate.
// Block = 256 threads = 8 waves; wave computes 16x64; block covers 128x64.
// The 64x32 W tile is identical for all 8 waves -> staged once per k-step in
// LDS (double buffered, one barrier per step). grid = (M/128, N/64).
// ---------------------------------------------------------------------------
#define WT_STRIDE 40   // 32 + 8 pad; 80B rows (16B multiple)

template <bool OUT_F32>
__global__ __launch_bounds__(256)
void gemm_bf16_nt(const unsigned short* __restrict__ A,
                  const unsigned short* __restrict__ W,
                  void* __restrict__ Cout, int M, int N, int K) {
    __shared__ unsigned short Wtile[2][64 * WT_STRIDE];
    int tid = threadIdx.x;
    int wave = tid >> 5, lane = tid & 31;
    int ln15 = lane & 15, hi = (lane >> 4) & 1;
    int m0 = blockIdx.x * 128 + wave * 16;
    int n0 = blockIdx.y * 64;
    size_t arow = (size_t)(m0 + ln15) * K;

    // cooperative staging: thread moves 8 contiguous bf16 (one b128)
    int sidx = tid * 8;                 // 0..2047
    int wr = sidx >> 5;                 // W row within tile (0..63)
    int wk = sidx & 31;                 // k within tile (multiple of 8)
    size_t wsrc = (size_t)(n0 + wr) * K + wk;
    int wdst = wr * WT_STRIDE + wk;

    v8f acc[4] = {};
    int buf = 0;
    for (int k0 = 0; k0 < K; k0 += 32, buf ^= 1) {
        // stage W tile (64 rows x 32 k)
        uint4 wv4 = *(const uint4*)&W[wsrc + k0];
        *(uint4*)&Wtile[buf][wdst] = wv4;

        // A fragment (16x32): lane = row, 16 consecutive bf16 per lane (k = 16*hi + s)
        BFrag a;
        int ka = k0 + hi * 16;
        a.q[0] = *(const uint4*)&A[arow + ka];
        a.q[1] = *(const uint4*)&A[arow + ka + 8];
        if (k0 + 160 <= K) __builtin_prefetch(&A[arow + ka + 128], 0, 0);

        __syncthreads();                // W tile visible to all waves

        // B fragments (32x16): lane = column (W row), 16 consecutive K
#pragma unroll
        for (int j = 0; j < 4; j++) {
            int lrow = (j * 16 + ln15) * WT_STRIDE + hi * 16;
            BFrag b;
            b.q[0] = *(const uint4*)&Wtile[buf][lrow];
            b.q[1] = *(const uint4*)&Wtile[buf][lrow + 8];
            acc[j] = __builtin_amdgcn_wmma_f32_16x16x32_bf16(
                false, a.v, false, b.v, (short)0, acc[j], false, false);
        }
    }

    int rb = m0 + hi * 8;
#pragma unroll
    for (int j = 0; j < 4; j++) {
        int col = n0 + j * 16 + ln15;
#pragma unroll
        for (int r = 0; r < 8; r++) {
            size_t off = (size_t)(rb + r) * N + col;
            if (OUT_F32) ((float*)Cout)[off] = acc[j][r];
            else ((unsigned short*)Cout)[off] = f2b(acc[j][r]);
        }
    }
}

// ---------------------------------------------------------------------------
// RoPE in-place on bf16 [BT, C] viewed as [BT, H, 64]; thread = (row, h, j<32)
// ---------------------------------------------------------------------------
__global__ void rope_inplace(unsigned short* __restrict__ qk) {
    int idx = blockIdx.x * blockDim.x + threadIdx.x;   // BT*H*32 threads
    int j = idx & 31;
    int rest = idx >> 5;
    int h = rest & (H_ - 1);
    int row = rest >> 4;
    int t = row & (T_ - 1);                            // position in sequence
    float ang = (float)t * __powf(10000.f, -(float)(2 * j) * (1.0f / 64.0f));
    float s, c;
    __sincosf(ang, &s, &c);
    size_t base = (size_t)row * C_ + h * D_;
    float q1 = b2f(qk[base + j]);
    float q2 = b2f(qk[base + j + 32]);
    qk[base + j]      = f2b(q1 * c - q2 * s);
    qk[base + j + 32] = f2b(q2 * c + q1 * s);
}

// ---------------------------------------------------------------------------
// V transpose: [BT, C] (heads interleaved) -> [B, H, D, T] bf16
// ---------------------------------------------------------------------------
__global__ void transpose_v(const unsigned short* __restrict__ Vb,
                            unsigned short* __restrict__ Vt) {
    int idx = blockIdx.x * blockDim.x + threadIdx.x;   // BT*C threads
    int c = idx & (C_ - 1);
    int row = idx >> 10;
    int s = row & (T_ - 1);
    int b = row >> 12;
    int h = c >> 6;
    int d = c & 63;
    Vt[((size_t)(b * H_ + h) * D_ + d) * T_ + s] = Vb[idx];
}

// ---------------------------------------------------------------------------
// Flash attention: block = 8 waves, all sharing (b,h); each wave owns a
// 16-query tile of a 128-query stripe. K (32x64) and V (64x32) tiles are
// identical across the block's waves -> staged once per 32-key step in LDS
// (double buffered, one barrier per step). Online softmax with 16-lane
// shuffles; P transposed C-layout -> A-layout via per-wave LDS.
// ---------------------------------------------------------------------------
#define KT_STRIDE 72   // 64 + 8 pad; 144B rows (16B multiple)
#define VT_STRIDE 40   // 32 + 8 pad; 80B rows (16B multiple)
#define PT_STRIDE 48   // 32 + 16 pad; 96B rows (16B multiple)

__global__ __launch_bounds__(256)
void attn_fwd(const unsigned short* __restrict__ Qb,
              const unsigned short* __restrict__ Kb,
              const unsigned short* __restrict__ Vt,
              unsigned short* __restrict__ AO) {
    __shared__ unsigned short Ktile[2][32 * KT_STRIDE];
    __shared__ unsigned short Vtile[2][64 * VT_STRIDE];
    __shared__ unsigned short Pbuf[8][16 * PT_STRIDE];

    int tid = threadIdx.x;
    int wave = tid >> 5, lane = tid & 31;
    int ln15 = lane & 15, hi = (lane >> 4) & 1;

    int h  = (blockIdx.x >> 5) & 15;
    int b  = blockIdx.x >> 9;
    int qs = (blockIdx.x & 31) * 128 + wave * 16;      // this wave's 16 queries
    unsigned short* pbuf = Pbuf[wave];

    // cooperative staging indices: thread moves one b128 of K and one of V
    int sidx = tid * 8;                                // 0..2047
    int krow = sidx >> 6;                              // key row 0..31
    int kcol = sidx & 63;                              // d, multiple of 8
    int vrow = sidx >> 5;                              // d row 0..63
    int vcol = sidx & 31;                              // s, multiple of 8
    size_t kgbase = ((size_t)b * T_ + krow) * C_ + h * D_ + kcol;   // + s0*C_
    size_t vgbase = (((size_t)b * H_ + h) * D_ + vrow) * T_ + vcol; // + s0
    int kdst = krow * KT_STRIDE + kcol;
    int vdst = vrow * VT_STRIDE + vcol;

    // Q fragments: A-layout, D split into two K=32 chunks
    BFrag aQ0, aQ1;
    size_t qrow = ((size_t)b * T_ + qs + ln15) * C_ + h * D_;
    int ka = hi * 16;
    aQ0.q[0] = *(const uint4*)&Qb[qrow + ka];
    aQ0.q[1] = *(const uint4*)&Qb[qrow + ka + 8];
    aQ1.q[0] = *(const uint4*)&Qb[qrow + 32 + ka];
    aQ1.q[1] = *(const uint4*)&Qb[qrow + 32 + ka + 8];

    v8f o0 = {}, o1 = {}, o2 = {}, o3 = {};
    float mrun[8], lrun[8];
#pragma unroll
    for (int r = 0; r < 8; r++) { mrun[r] = -3.0e38f; lrun[r] = 0.f; }

    int buf = 0;
    for (int s0 = 0; s0 < T_; s0 += 32, buf ^= 1) {
        // ---- stage K/V tiles for this 32-key step ----
        uint4 kx = *(const uint4*)&Kb[kgbase + (size_t)s0 * C_];
        uint4 vx = *(const uint4*)&Vt[vgbase + s0];
        *(uint4*)&Ktile[buf][kdst] = kx;
        *(uint4*)&Vtile[buf][vdst] = vx;
        __syncthreads();

        // ---- scores: S[16q x 32k] = Q(16x64) * K^T(64x32) ----
        v8f S0 = {}, S1 = {};
        BFrag bk;
        int kl0 = ln15 * KT_STRIDE + hi * 16;          // key row ln15
        bk.q[0] = *(const uint4*)&Ktile[buf][kl0];
        bk.q[1] = *(const uint4*)&Ktile[buf][kl0 + 8];
        S0 = __builtin_amdgcn_wmma_f32_16x16x32_bf16(false, aQ0.v, false, bk.v, (short)0, S0, false, false);
        bk.q[0] = *(const uint4*)&Ktile[buf][kl0 + 32];
        bk.q[1] = *(const uint4*)&Ktile[buf][kl0 + 40];
        S0 = __builtin_amdgcn_wmma_f32_16x16x32_bf16(false, aQ1.v, false, bk.v, (short)0, S0, false, false);
        int kl1 = kl0 + 16 * KT_STRIDE;                // key row ln15+16
        bk.q[0] = *(const uint4*)&Ktile[buf][kl1];
        bk.q[1] = *(const uint4*)&Ktile[buf][kl1 + 8];
        S1 = __builtin_amdgcn_wmma_f32_16x16x32_bf16(false, aQ0.v, false, bk.v, (short)0, S1, false, false);
        bk.q[0] = *(const uint4*)&Ktile[buf][kl1 + 32];
        bk.q[1] = *(const uint4*)&Ktile[buf][kl1 + 40];
        S1 = __builtin_amdgcn_wmma_f32_16x16x32_bf16(false, aQ1.v, false, bk.v, (short)0, S1, false, false);

        // ---- online softmax (row = fixed acc reg across a 16-lane group) ----
        float alpha[8];
#pragma unroll
        for (int r = 0; r < 8; r++) {
            float a0 = S0[r] * 0.125f, a1 = S1[r] * 0.125f;  // 1/sqrt(64)
            float tmax = fmaxf(a0, a1);
            tmax = fmaxf(tmax, __shfl_xor(tmax, 1, 32));
            tmax = fmaxf(tmax, __shfl_xor(tmax, 2, 32));
            tmax = fmaxf(tmax, __shfl_xor(tmax, 4, 32));
            tmax = fmaxf(tmax, __shfl_xor(tmax, 8, 32));
            float mn = fmaxf(mrun[r], tmax);
            float al = __expf(mrun[r] - mn);
            float p0 = __expf(a0 - mn);
            float p1 = __expf(a1 - mn);
            float rs = p0 + p1;
            rs += __shfl_xor(rs, 1, 32);
            rs += __shfl_xor(rs, 2, 32);
            rs += __shfl_xor(rs, 4, 32);
            rs += __shfl_xor(rs, 8, 32);
            mrun[r] = mn;
            lrun[r] = lrun[r] * al + rs;
            alpha[r] = al;
            // C-layout -> LDS (row-major 16 x 32 with pad)
            pbuf[(r + hi * 8) * PT_STRIDE + ln15]      = f2b(p0);
            pbuf[(r + hi * 8) * PT_STRIDE + ln15 + 16] = f2b(p1);
        }
        asm volatile("s_wait_dscnt 0" ::: "memory");

        // ---- reload P as A-fragment (16x32) ----
        BFrag ap;
        ap.q[0] = *(const uint4*)&pbuf[ln15 * PT_STRIDE + hi * 16];
        ap.q[1] = *(const uint4*)&pbuf[ln15 * PT_STRIDE + hi * 16 + 8];

        // ---- rescale O, accumulate P @ V (keys s0..s0+31) ----
#pragma unroll
        for (int r = 0; r < 8; r++) {
            o0[r] *= alpha[r]; o1[r] *= alpha[r];
            o2[r] *= alpha[r]; o3[r] *= alpha[r];
        }
        BFrag bv;
        int vl = ln15 * VT_STRIDE + hi * 16;           // d row = j*16 + ln15
        bv.q[0] = *(const uint4*)&Vtile[buf][vl];
        bv.q[1] = *(const uint4*)&Vtile[buf][vl + 8];
        o0 = __builtin_amdgcn_wmma_f32_16x16x32_bf16(false, ap.v, false, bv.v, (short)0, o0, false, false);
        vl += 16 * VT_STRIDE;
        bv.q[0] = *(const uint4*)&Vtile[buf][vl];
        bv.q[1] = *(const uint4*)&Vtile[buf][vl + 8];
        o1 = __builtin_amdgcn_wmma_f32_16x16x32_bf16(false, ap.v, false, bv.v, (short)0, o1, false, false);
        vl += 16 * VT_STRIDE;
        bv.q[0] = *(const uint4*)&Vtile[buf][vl];
        bv.q[1] = *(const uint4*)&Vtile[buf][vl + 8];
        o2 = __builtin_amdgcn_wmma_f32_16x16x32_bf16(false, ap.v, false, bv.v, (short)0, o2, false, false);
        vl += 16 * VT_STRIDE;
        bv.q[0] = *(const uint4*)&Vtile[buf][vl];
        bv.q[1] = *(const uint4*)&Vtile[buf][vl + 8];
        o3 = __builtin_amdgcn_wmma_f32_16x16x32_bf16(false, ap.v, false, bv.v, (short)0, o3, false, false);
    }

    // ---- normalize and write bf16 attention output [BT, C] ----
    size_t orow = ((size_t)b * T_ + qs + hi * 8) * C_ + h * D_;
#pragma unroll
    for (int r = 0; r < 8; r++) {
        float inv = 1.0f / lrun[r];
        AO[orow + (size_t)r * C_ + ln15]      = f2b(o0[r] * inv);
        AO[orow + (size_t)r * C_ + 16 + ln15] = f2b(o1[r] * inv);
        AO[orow + (size_t)r * C_ + 32 + ln15] = f2b(o2[r] * inv);
        AO[orow + (size_t)r * C_ + 48 + ln15] = f2b(o3[r] * inv);
    }
}

// ---------------------------------------------------------------------------
// Launch
// ---------------------------------------------------------------------------
extern "C" void kernel_launch(void* const* d_in, const int* in_sizes, int n_in,
                              void* d_out, int out_size, void* d_ws, size_t ws_size,
                              hipStream_t stream) {
    const float* x  = (const float*)d_in[0];
    const float* wq = (const float*)d_in[1];
    const float* wk = (const float*)d_in[2];
    const float* wv = (const float*)d_in[3];
    const float* wo = (const float*)d_in[4];
    float* out = (float*)d_out;

    unsigned char* ws = (unsigned char*)d_ws;
    const size_t MB = (size_t)1 << 20;
    unsigned short* xb  = (unsigned short*)(ws + 0 * MB);   // 16 MB
    unsigned short* wqb = (unsigned short*)(ws + 16 * MB);  // 2 MB
    unsigned short* wkb = (unsigned short*)(ws + 18 * MB);
    unsigned short* wvb = (unsigned short*)(ws + 20 * MB);
    unsigned short* wob = (unsigned short*)(ws + 22 * MB);
    unsigned short* Qb  = (unsigned short*)(ws + 24 * MB);  // 16 MB each
    unsigned short* Kb  = (unsigned short*)(ws + 40 * MB);
    unsigned short* Vb  = (unsigned short*)(ws + 56 * MB);
    unsigned short* Vt  = (unsigned short*)(ws + 72 * MB);
    unsigned short* AO  = (unsigned short*)(ws + 88 * MB);  // total 104 MB

    const int nX = BT_ * C_;       // 8388608
    const int nW = C_ * C_;        // 1048576
    cvt_f32_bf16<<<(nX + 255) / 256, 256, 0, stream>>>(x,  xb,  nX);
    cvt_f32_bf16<<<(nW + 255) / 256, 256, 0, stream>>>(wq, wqb, nW);
    cvt_f32_bf16<<<(nW + 255) / 256, 256, 0, stream>>>(wk, wkb, nW);
    cvt_f32_bf16<<<(nW + 255) / 256, 256, 0, stream>>>(wv, wvb, nW);
    cvt_f32_bf16<<<(nW + 255) / 256, 256, 0, stream>>>(wo, wob, nW);

    dim3 gg(BT_ / 128, C_ / 64);   // (64, 16)
    gemm_bf16_nt<false><<<gg, 256, 0, stream>>>(xb, wqb, Qb, BT_, C_, C_);
    gemm_bf16_nt<false><<<gg, 256, 0, stream>>>(xb, wkb, Kb, BT_, C_, C_);
    gemm_bf16_nt<false><<<gg, 256, 0, stream>>>(xb, wvb, Vb, BT_, C_, C_);

    const int nR = BT_ * H_ * 32;  // 4194304
    rope_inplace<<<nR / 256, 256, 0, stream>>>(Qb);
    rope_inplace<<<nR / 256, 256, 0, stream>>>(Kb);

    transpose_v<<<nX / 256, 256, 0, stream>>>(Vb, Vt);

    attn_fwd<<<1024, 256, 0, stream>>>(Qb, Kb, Vt, AO);   // 2*16*32 blocks

    gemm_bf16_nt<true><<<gg, 256, 0, stream>>>(AO, wob, out, BT_, C_, C_);
}
